// VAE_45535243272424
// MI455X (gfx1250) — matrix operand order repaired and checked by reading, not compiled, not run on hardware
//
#include <hip/hip_runtime.h>

// ---------------------------------------------------------------------------
// VAE total-loss fused kernel for MI455X (gfx1250, wave32).
//
// Roofline: ~120 MB of f32 input, 1 float out -> ~5us at 23.3 TB/s. Purely
// bandwidth bound; design = one streaming pass per row, all row-local state in
// LDS, WMMA (f32_16x16x32_f16) used for the per-row adjacent dot products
// (V=28 pads into K=32 exactly; we consume the D-matrix diagonal).
// Round 2: branchless cndmask diagonal extraction (no scratch spill).
// Round 3: zeroed pad column + clamped LDS indices so all fragment loads are
// unconditional (kills the divergent saveexec/ds_load boundary blocks).
// ---------------------------------------------------------------------------

typedef __attribute__((ext_vector_type(16))) _Float16 v16h;
typedef __attribute__((ext_vector_type(8)))  float    v8f;

#define NB      8192
#define NL      64
#define NV      28
#define LATENT  32
#define PAD_TOK 26
#define EOS_TOK 27
#define EPSF    1e-7f
#define RSTRIDE 29              // LDS row stride (odd -> conflict-free columns)

// workspace accumulator slots (floats)
#define WS_REC    0
#define WS_MASK   1
#define WS_CONSEC 2
#define WS_CNTEXP 3
#define WS_KL     4
#define WS_CLS    5
#define WS_COL    6      // 28 entries
#define WS_TOTAL  34

__device__ __constant__ float c_freqs[NV] = {
  0.0817f,0.0149f,0.0278f,0.0425f,0.127f,0.0223f,0.0202f,0.0609f,0.0697f,
  0.0015f,0.0077f,0.0403f,0.0241f,0.0675f,0.0751f,0.0193f,0.001f,0.0599f,
  0.0633f,0.0906f,0.0276f,0.0098f,0.0236f,0.0015f,0.0197f,0.0007f,0.f,0.f};

__device__ inline float wave_sum32(float v) {
#pragma unroll
  for (int o = 16; o > 0; o >>= 1) v += __shfl_xor(v, o, 32);
  return v;
}

// Branchless extract of element i (0..7) from a v8f held in VGPRs.
// Constant indices only -> lowers to a v_cndmask chain, no scratch spill.
__device__ inline float pick8_sel(v8f c, int i) {
  float v = c[0];
  v = (i == 1) ? c[1] : v;
  v = (i == 2) ? c[2] : v;
  v = (i == 3) ? c[3] : v;
  v = (i == 4) ? c[4] : v;
  v = (i == 5) ? c[5] : v;
  v = (i == 6) ? c[6] : v;
  v = (i == 7) ? c[7] : v;
  return v;
}

__global__ void init_ws(float* ws) {
  int i = threadIdx.x;
  if (i < WS_TOTAL) ws[i] = 0.f;
}

// One wave32 per row b. 8 waves / block, 1024 blocks = 8192 rows exactly.
__global__ __launch_bounds__(256) void row_kernel(
    const float* __restrict__ x,
    const float* __restrict__ recon,
    float* __restrict__ ws)
{
  __shared__ float sR[8][NL * RSTRIDE];  // recon row; col 28 is a zero pad
  __shared__ float sAdj[8][NL];          // adjacent dot products (63 used)
  __shared__ int   sTgt[8][NL];          // argmax targets
  __shared__ float sW[NL];               // i^2.5 weights

  const int tid  = threadIdx.x;
  const int w    = tid >> 5;
  const int lane = tid & 31;
  const int b    = blockIdx.x * 8 + w;

  if (tid < NL) sW[tid] = powf((float)tid, 2.5f);

  // ---- stage recon row into LDS (coalesced stream, stride-29 layout) ----
  const float* rrow = recon + (size_t)b * (NL * NV);
  for (int i = lane; i < NL * NV; i += 32) {
    int r = i / NV, c = i - r * NV;
    sR[w][r * RSTRIDE + c] = rrow[i];
  }
  // zero pad column 28: clamped reads of K=28..31 return 0 with no guards
  sR[w][lane * RSTRIDE + NV] = 0.f;
  sR[w][(lane + 32) * RSTRIDE + NV] = 0.f;
  __syncthreads();

  // ---- adjacent[j] = dot(R[j+1,:], R[j,:]) via 4x v_wmma_f32_16x16x32_f16 ----
  // D[m][n] = sum_k A[m][k]*B[k][n], A row m = R[j0+1+m][k], B[k][n] = R[j0+n][k]
  // -> diagonal D[i][i] = adjacent[j0+i]. K=32 covers V=28 (K>=28 hits pad col).
  const int m  = lane & 15;
  const int hi = lane >> 4;
  // diagonal ownership (uniform per lane across tiles):
  const bool isDiag = (lane < 8) || (lane >= 24);
  const int  dOff   = (lane < 8) ? lane : (lane - 16);  // d index within tile
#pragma unroll
  for (int tile = 0; tile < 4; ++tile) {
    const int j0 = tile * 16;
    v16h af, bf;
    // A fragment: 16-bit A 16x32 layout. lane holds row m; VGPR v:
    //   k = (v>=4)*16 + hi*8 + (v&3)*2  (pair k, k+1)
    const int rowA   = j0 + 1 + m;
    const bool rowOK = (rowA < NL);              // only tile 3, m==15 fails
    const int rowAc  = rowOK ? rowA : (NL - 1);  // clamped, always-legal address
#pragma unroll
    for (int v = 0; v < 8; ++v) {
      int kb = ((v >> 2) << 4) + (hi << 3) + ((v & 3) << 1);
      int k0 = (kb     < NV) ? kb     : NV;      // clamp into zero pad col
      int k1 = (kb + 1 < NV) ? kb + 1 : NV;
      float a0 = sR[w][rowAc * RSTRIDE + k0];
      float a1 = sR[w][rowAc * RSTRIDE + k1];
      a0 = rowOK ? a0 : 0.f;
      a1 = rowOK ? a1 : 0.f;
      af[2 * v]     = (_Float16)a0;
      af[2 * v + 1] = (_Float16)a1;
    }
    // B fragment: 32x16, same k order as A: VGPR v -> K=(v>=4)*16+hi*8+(v&3)*2 (+1);
    // column n = lane&15 -> row R[j0+n] (always <64).
    const int rowB = j0 + m;
    const int koff = hi << 4;
#pragma unroll
    for (int v = 0; v < 8; ++v) {
      int k  = ((v >> 2) << 4) + (koff >> 1) + ((v & 3) << 1);
      int k0 = (k     < NV) ? k     : NV;
      int k1 = (k + 1 < NV) ? k + 1 : NV;
      bf[2 * v]     = (_Float16)sR[w][rowB * RSTRIDE + k0];
      bf[2 * v + 1] = (_Float16)sR[w][rowB * RSTRIDE + k1];
    }
    v8f c = {};
    c = __builtin_amdgcn_wmma_f32_16x16x32_f16(
        /*neg_a=*/false, af, /*neg_b=*/false, bf,
        /*c_mod=*/(short)0, c, /*reuse_a=*/false, /*reuse_b=*/false);
    // C/D layout: lanes 0-15 N=lane, VGPR r -> M=r; lanes 16-31 N=lane-16, M=r+8.
    // Diagonal holders: lanes 0-7 (elem=lane, d=lane), lanes 24-31 (elem=lane-24,
    // d=lane-16). Extract with constant-index selects (all lanes), store predicated.
    float dv = pick8_sel(c, lane & 7);
    if (isDiag) sAdj[w][j0 + dOff] = dv;
  }

  // ---- targets = argmax_v x, reconstruction CE, masks (2 positions/lane) ----
  const float* xrow = x + (size_t)b * (NL * NV);
  float recAcc = 0.f, maskAcc = 0.f, ceAcc = 0.f;
#pragma unroll
  for (int q = 0; q < 2; ++q) {
    int l = lane + 32 * q;
    const float* xp = xrow + l * NV;
    int best = 0; float bv = xp[0];
#pragma unroll
    for (int v = 1; v < NV; ++v) {
      float xv = xp[v];
      if (xv > bv) { bv = xv; best = v; }   // first-occurrence argmax
    }
    sTgt[w][l] = best;
    float mask = (best != PAD_TOK) ? 1.f : 0.f;
    float p = sR[w][l * RSTRIDE + best];
    p = fminf(fmaxf(p, EPSF), 1.0f);
    recAcc  += -logf(p) * mask;
    maskAcc += mask;
    ceAcc   += (best != PAD_TOK && best != EOS_TOK) ? 1.f : 0.f;
  }
  __syncthreads();

  // ---- apply adjacency mask: adjacent[j] *= (tgt[j+1] != PAD) ----
#pragma unroll
  for (int q = 0; q < 2; ++q) {
    int j = lane + 32 * q;
    if (j < NL - 1) {
      if (sTgt[w][j + 1] == PAD_TOK) sAdj[w][j] = 0.f;
    } else if (j == NL - 1) {
      sAdj[w][NL - 1] = 0.f;                // unused slot
    }
  }
  __syncthreads();

  // ---- consecutive similarity: sum_i i^2.5 * sum_j prod(adj[j..j+i-1]).
  // Reordered per-lane: lane j accumulates sum_k w[k+1]*p_k[j] locally. ----
  float cons = 0.f;
#pragma unroll
  for (int q = 0; q < 2; ++q) {
    int j = lane + 32 * q;
    if (j < NL - 1) {
      float p = sAdj[w][j];
      float s = p;                          // i=1, weight 1^2.5
      for (int k = 1; j + k < NL - 1; ++k) {
        p *= sAdj[w][j + k];
        if (p == 0.f) break;                // masked window -> stays 0
        s += sW[k + 1] * p;
      }
      cons += s;
    }
  }

  // ---- frequency column sums: lane v accumulates sum_l maskexp[l]*R[l][v] ----
  float fs = 0.f;
  if (lane < NV) {
    for (int l = 0; l < NL; ++l) {
      int t = sTgt[w][l];                   // broadcast read
      if (t != PAD_TOK && t != EOS_TOK) fs += sR[w][l * RSTRIDE + lane];
    }
  }

  // ---- per-wave reductions + global accumulation ----
  recAcc  = wave_sum32(recAcc);
  maskAcc = wave_sum32(maskAcc);
  ceAcc   = wave_sum32(ceAcc);
  cons    = wave_sum32(cons);
  if (lane == 0) {
    atomicAdd(&ws[WS_REC],    recAcc);
    atomicAdd(&ws[WS_MASK],   maskAcc);
    atomicAdd(&ws[WS_CNTEXP], ceAcc);
    atomicAdd(&ws[WS_CONSEC], cons);
  }
  if (lane < NV) atomicAdd(&ws[WS_COL + lane], fs);
}

// KL + classifier partial sums (tiny tensors).
__global__ __launch_bounds__(256) void kl_cls_kernel(
    const float* __restrict__ zm, const float* __restrict__ zlv,
    const float* __restrict__ cs, float* __restrict__ ws)
{
  const int gid    = blockIdx.x * blockDim.x + threadIdx.x;
  const int stride = gridDim.x * blockDim.x;
  float kl = 0.f, cl = 0.f;
  for (int i = gid; i < NB * LATENT; i += stride) {
    float mu = zm[i], lv = zlv[i];
    kl += 1.f + lv - mu * mu - expf(lv);
  }
  for (int i = gid; i < NB; i += stride) cl += logf(cs[i] + EPSF);
  kl = wave_sum32(kl);
  cl = wave_sum32(cl);
  if ((threadIdx.x & 31) == 0) {
    atomicAdd(&ws[WS_KL],  kl);
    atomicAdd(&ws[WS_CLS], cl);
  }
}

__global__ void finalize_kernel(const float* __restrict__ ws, float* __restrict__ out) {
  float rec  = ws[WS_REC] / ws[WS_MASK];
  float kl   = -0.5f * (ws[WS_KL] / (float)(NB * LATENT));
  float cls  = -(ws[WS_CLS] / (float)NB);
  float cons = ws[WS_CONSEC];
  float cnt  = ws[WS_CNTEXP];
  float freq = 0.f;
  for (int v = 0; v < NV; ++v) {
    float mlp = ws[WS_COL + v] / cnt;
    freq += fmaxf(0.f, mlp - c_freqs[v]);
  }
  out[0] = rec + 0.1f * kl + 2.0f * cls + 0.2f * cons + 0.5f * freq;
}

extern "C" void kernel_launch(void* const* d_in, const int* in_sizes, int n_in,
                              void* d_out, int out_size, void* d_ws, size_t ws_size,
                              hipStream_t stream) {
  (void)in_sizes; (void)n_in; (void)out_size; (void)ws_size;
  const float* x     = (const float*)d_in[0];
  const float* recon = (const float*)d_in[1];
  const float* zm    = (const float*)d_in[2];
  const float* zlv   = (const float*)d_in[3];
  const float* cs    = (const float*)d_in[4];
  float* ws  = (float*)d_ws;
  float* out = (float*)d_out;

  hipLaunchKernelGGL(init_ws,         dim3(1),    dim3(64),  0, stream, ws);
  hipLaunchKernelGGL(row_kernel,      dim3(1024), dim3(256), 0, stream, x, recon, ws);
  hipLaunchKernelGGL(kl_cls_kernel,   dim3(256),  dim3(256), 0, stream, zm, zlv, cs, ws);
  hipLaunchKernelGGL(finalize_kernel, dim3(1),    dim3(1),   0, stream, ws, out);
}
